// DTIN_45226005627020
// MI455X (gfx1250) — matrix operand order repaired and checked by reading, not compiled, not run on hardware
//
#include <hip/hip_runtime.h>
#include <hip/hip_bf16.h>

// ---------------------------------------------------------------------------
// MI455X (gfx1250) Conv1x1+BN+SiLU -> AttnDyConv pipeline.
// bf16 WMMA (v_wmma_f32_16x16x32_bf16) for both GEMM-shaped convs.
// k_dyconv: double-buffered LDS staging of B tiles, software-pipelined copy,
// XOR-swizzled B layout (bank-conflict-free ds_load_b128), register
// double-buffered LDS reads. k_conv1x1: depth-2 register pipeline on B.
// ---------------------------------------------------------------------------

typedef __attribute__((ext_vector_type(16))) __bf16 v16bf;
typedef __attribute__((ext_vector_type(8)))  __bf16 v8bf;
typedef __attribute__((ext_vector_type(8)))  float  v8f;
typedef __attribute__((ext_vector_type(4)))  float  v4f;

#define EPSV 1e-5f

#define BN_  16
#define C1_  128
#define C2_  256
#define HW_  6400   // 80*80
#define WD_  80
#define PW_  82     // padded width/height

__device__ __forceinline__ v8f wmma_bf16(v16bf a, v16bf b, v8f c) {
#if defined(__AMDGCN__)
    return __builtin_amdgcn_wmma_f32_16x16x32_bf16(false, a, false, b,
                                                   (short)0, c, false, false);
#else
    (void)a; (void)b;
    return c;
#endif
}

__device__ __forceinline__ __bf16 f2bf(float f) { return (__bf16)f; }

// --------------------------------------------------------------------- zero
__global__ void k_zero(uint4* p, long long n4) {
    long long i = (long long)blockIdx.x * blockDim.x + threadIdx.x;
    long long stride = (long long)gridDim.x * blockDim.x;
    uint4 z; z.x = 0u; z.y = 0u; z.z = 0u; z.w = 0u;
    for (; i < n4; i += stride) p[i] = z;
}

// ------------------------------------------------- x NCHW f32 -> NHWC bf16
__global__ void k_convert_x(const float* __restrict__ x, __bf16* __restrict__ xn) {
    long long i = (long long)blockIdx.x * 256 + threadIdx.x;
    const long long total = (long long)BN_ * C1_ * HW_;
    if (i >= total) return;
    int p = (int)(i % HW_);
    int c = (int)((i / HW_) % C1_);
    int b = (int)(i / ((long long)HW_ * C1_));
    xn[((long long)b * HW_ + p) * C1_ + c] = f2bf(x[i]);
}

// ---------------------- pack conv1_w[o][c] into WMMA B-operand lane layout
// layout: [ot(16)][c32(4)][lane(32)][j(8)] dwords, dword = bf16x2(K=2j,2j+1)
__global__ void k_pack_w1(const float* __restrict__ w1, unsigned* __restrict__ w1pk) {
    int i = blockIdx.x * 256 + threadIdx.x;
    if (i >= 16 * 4 * 32 * 8) return;
    int j    = i & 7;
    int lane = (i >> 3) & 31;
    int c32  = (i >> 8) & 3;
    int ot   = i >> 10;
    int n  = lane & 15;
    int kk = 2 * j + ((lane & 16) ? 8 : 0) + ((j & 4) ? 8 : 0);
    int o  = ot * 16 + n;
    int c  = c32 * 32 + kk;
    union { __bf16 h[2]; unsigned u; } u;
    u.h[0] = f2bf(w1[o * C1_ + c]);
    u.h[1] = f2bf(w1[o * C1_ + c + 1]);
    w1pk[i] = u.u;
}

// ---------------------------------- 1x1 conv (WMMA) + BN + SiLU + stats
__global__ __launch_bounds__(256) void k_conv1x1(
    const __bf16* __restrict__ xn, const unsigned* __restrict__ w1pk,
    const float* __restrict__ bng, const float* __restrict__ bnb,
    const float* __restrict__ bnm, const float* __restrict__ bnv,
    __bf16* __restrict__ y, float* __restrict__ gsum, float* __restrict__ gsq)
{
    __shared__ float lsum[C2_];
    __shared__ float lsq[C2_];
    int tid = threadIdx.x;
    lsum[tid] = 0.0f; lsq[tid] = 0.0f;
    __syncthreads();

    int b    = blockIdx.y;
    int wave = tid >> 5, lane = tid & 31;
    int ptile = blockIdx.x * 8 + wave;      // 0..399
    int p0 = ptile * 16;
    int m    = lane & 15;
    int koff = (lane & 16) ? 8 : 0;

    v8f zero8;
#pragma unroll
    for (int r = 0; r < 8; r++) zero8[r] = 0.0f;
    v8f acc[16];
#pragma unroll
    for (int ot = 0; ot < 16; ot++) acc[ot] = zero8;

    const __bf16* abase = xn + ((long long)b * HW_ + p0 + m) * C1_ + koff;

#pragma unroll
    for (int c32 = 0; c32 < 4; c32++) {
        union { v16bf v; v8bf h[2]; } A;
        A.h[0] = *(const v8bf*)(abase + c32 * 32);
        A.h[1] = *(const v8bf*)(abase + c32 * 32 + 16);
        // depth-2 register pipeline: two B-tile pairs in flight during WMMA
        v16bf B0 = *(const v16bf*)(w1pk + (((0 * 4 + c32) * 32 + lane) << 3));
        v16bf B1 = *(const v16bf*)(w1pk + (((1 * 4 + c32) * 32 + lane) << 3));
#pragma unroll
        for (int ot = 0; ot < 16; ot++) {
            int otn = (ot + 2) & 15;
            v16bf B2 = *(const v16bf*)(w1pk + (((otn * 4 + c32) * 32 + lane) << 3));
            acc[ot] = wmma_bf16(A.v, B0, acc[ot]);
            B0 = B1;
            B1 = B2;
        }
    }

    int n    = lane & 15;
    int prow = (lane & 16) ? 8 : 0;
#pragma unroll
    for (int ot = 0; ot < 16; ot++) {
        int o = ot * 16 + n;
        float s    = bng[o] * rsqrtf(bnv[o] + EPSV);
        float bias = bnb[o] - bnm[o] * s;
        float ps = 0.0f, pq = 0.0f;
#pragma unroll
        for (int r = 0; r < 8; r++) {
            float v = acc[ot][r] * s + bias;
            v = v / (1.0f + __expf(-v));          // SiLU
            int p = p0 + prow + r;
            y[((long long)b * HW_ + p) * C2_ + o] = f2bf(v);
            ps += v; pq += v * v;
        }
        atomicAdd(&lsum[o], ps);                   // ds_add_f32
        atomicAdd(&lsq[o],  pq);
    }
    __syncthreads();
    atomicAdd(&gsum[b * C2_ + tid], lsum[tid]);    // global_atomic_add_f32
    atomicAdd(&gsq[b * C2_ + tid],  lsq[tid]);
}

// --------------------------------------------- per-(b,c) mean / rstd finalize
__global__ void k_finalize(const float* __restrict__ gsum, const float* __restrict__ gsq,
                           float* __restrict__ mu, float* __restrict__ rstd) {
    int i = blockIdx.x * 256 + threadIdx.x;
    if (i >= BN_ * C2_) return;
    float m = gsum[i] * (1.0f / (float)HW_);
    float v = gsq[i] * (1.0f / (float)HW_) - m * m;
    mu[i] = m;
    rstd[i] = rsqrtf(v + EPSV);
}

// ------------------------------- attention: fc1 -> relu -> fc2 -> softmax(K=4)
__global__ void k_attn(const float* __restrict__ mu,
                       const float* __restrict__ fc1w, const float* __restrict__ fc1b,
                       const float* __restrict__ fc2w, const float* __restrict__ fc2b,
                       float* __restrict__ attn) {
    __shared__ float red[4][64];
    int b = blockIdx.x, t = threadIdx.x;
    float part[4] = {0.f, 0.f, 0.f, 0.f};
    for (int c = t; c < C2_; c += 64) {
        float pv = mu[b * C2_ + c];
#pragma unroll
        for (int k = 0; k < 4; k++) part[k] += pv * fc1w[k * C2_ + c];
    }
#pragma unroll
    for (int k = 0; k < 4; k++) red[k][t] = part[k];
    __syncthreads();
    if (t < 4) {
        float a = 0.0f;
        for (int i = 0; i < 64; i++) a += red[t][i];
        a += fc1b[t];
        red[t][0] = fmaxf(a, 0.0f);               // relu
    }
    __syncthreads();
    if (t == 0) {
        float a0 = red[0][0], a1 = red[1][0], a2 = red[2][0], a3 = red[3][0];
        float z[4];
#pragma unroll
        for (int k = 0; k < 4; k++)
            z[k] = fc2b[k] + a0 * fc2w[k * 4 + 0] + a1 * fc2w[k * 4 + 1]
                           + a2 * fc2w[k * 4 + 2] + a3 * fc2w[k * 4 + 3];
        float mx = fmaxf(fmaxf(z[0], z[1]), fmaxf(z[2], z[3]));
        float e[4], s = 0.0f;
#pragma unroll
        for (int k = 0; k < 4; k++) { e[k] = __expf(z[k] - mx); s += e[k]; }
#pragma unroll
        for (int k = 0; k < 4; k++) attn[b * 4 + k] = e[k] / s;
    }
}

// --------- aggregate dy_w over K with attn, pack into WMMA B lane layout
// layout: [b][t(9)][c32(8)][ot(16)][lane(32)][jj(8)] dwords (bf16x2 K-pairs)
// XOR granule swizzle: physical 16B-granule = logical granule ^ ((lane>>3)&1)
// -> each wave-wide ds_load_b128 in k_dyconv hits all 64 LDS banks exactly 2x.
__global__ void k_agg(const float* __restrict__ dyw, const float* __restrict__ attn,
                      unsigned* __restrict__ aggpk) {
    long long i = (long long)blockIdx.x * 256 + threadIdx.x;
    const long long total = (long long)BN_ * 9 * 8 * 16 * 32 * 8;
    if (i >= total) return;
    int j    = (int)(i & 7);
    int lane = (int)(i >> 3) & 31;
    int ot   = (int)(i >> 8) & 15;
    int c32  = (int)(i >> 12) & 7;
    long long rest = i >> 15;
    int t = (int)(rest % 9);
    int b = (int)(rest / 9);

    int n  = lane & 15;
    int kk = 2 * j + ((lane & 16) ? 8 : 0) + ((j & 4) ? 8 : 0);
    int o  = ot * 16 + n;
    int ii = c32 * 32 + kk;

    float a0 = attn[b * 4 + 0], a1 = attn[b * 4 + 1];
    float a2 = attn[b * 4 + 2], a3 = attn[b * 4 + 3];
    const long long ks = (long long)C2_ * C2_ * 9;     // per-k stride in dy_w
    long long base = ((long long)o * C2_ + ii) * 9 + t;

    float w0 = a0 * dyw[base]          + a1 * dyw[base + ks]
             + a2 * dyw[base + 2 * ks] + a3 * dyw[base + 3 * ks];
    long long base2 = base + 9;                         // next input channel
    float w1 = a0 * dyw[base2]          + a1 * dyw[base2 + ks]
             + a2 * dyw[base2 + 2 * ks] + a3 * dyw[base2 + 3 * ks];

    union { __bf16 h[2]; unsigned u; } u;
    u.h[0] = f2bf(w0); u.h[1] = f2bf(w1);

    int p2 = (lane >> 3) & 1;                           // bank swizzle
    int jj = (j & 3) | (((j >> 2) ^ p2) << 2);
    aggpk[(i & ~7LL) | jj] = u.u;
}

// -------------------- instance-normalize y, write into padded NHWC bf16
__global__ void k_normpad(const __bf16* __restrict__ y, const float* __restrict__ mu,
                          const float* __restrict__ rstd, __bf16* __restrict__ ynp) {
    int b = blockIdx.y, p = blockIdx.x, c = threadIdx.x;
    int h = p / WD_, w = p % WD_;
    float v = (float)y[((long long)b * HW_ + p) * C2_ + c];
    v = (v - mu[b * C2_ + c]) * rstd[b * C2_ + c];
    ynp[(((long long)b * PW_ + (h + 1)) * PW_ + (w + 1)) * C2_ + c] = f2bf(v);
}

// ------------------- main dynamic 3x3 conv (implicit GEMM, bf16 WMMA) + bn1
// grid (50, 16), block 256 = 8 waves; wave -> one 16-pixel stripe, all 256 oc.
// B tiles double-buffered through LDS: global loads for stage s+1 are issued
// before the WMMA block of stage s and committed to LDS afterwards.
__global__ __launch_bounds__(256) void k_dyconv(
    const __bf16* __restrict__ ynp, const unsigned* __restrict__ aggpk,
    const float* __restrict__ attn, const float* __restrict__ dyb,
    const float* __restrict__ b1g, const float* __restrict__ b1b,
    const float* __restrict__ b1m, const float* __restrict__ b1v,
    float* __restrict__ out)
{
    __shared__ __align__(32) unsigned Bsm[2][4096];   // 2 x 16KB weight stages

    int tid  = threadIdx.x;
    int b    = blockIdx.y;
    int wave = tid >> 5, lane = tid & 31;
    int ptile = blockIdx.x * 8 + wave;      // 0..399
    int p0 = ptile * 16;
    int h = p0 / WD_, w0 = p0 % WD_;        // 80 % 16 == 0: stripe stays in-row
    int m    = lane & 15;
    int koff = (lane & 16) ? 8 : 0;
    int swz  = (lane >> 3) & 1;             // granule swizzle (matches k_agg)

    v8f zero8;
#pragma unroll
    for (int r = 0; r < 8; r++) zero8[r] = 0.0f;
    v8f acc[16];
#pragma unroll
    for (int ot = 0; ot < 16; ot++) acc[ot] = zero8;

    const unsigned* bpk = aggpk + (long long)b * 9 * 8 * 16 * 32 * 8;

    // prologue: stage 0 -> LDS buffer 0
    {
        const uint4* src = (const uint4*)bpk;           // t=0,c32=0
        uint4* dst = (uint4*)Bsm[0];
#pragma unroll
        for (int q = 0; q < 4; q++) dst[tid + q * 256] = src[tid + q * 256];
    }
    __syncthreads();

    for (int s = 0; s < 72; s++) {              // s = t*8 + c32
        int t = s >> 3, c32 = s & 7;
        int kh = t / 3, kw = t % 3;

        // ---- issue next stage's global loads (kept in regs during compute)
        uint4 c0, c1, c2, c3;
        if (s + 1 < 72) {
            int tn = (s + 1) >> 3, cn = (s + 1) & 7;
            const uint4* src = (const uint4*)(bpk + (long long)tn * 32768 + cn * 4096);
            c0 = src[tid];
            c1 = src[tid + 256];
            c2 = src[tid + 512];
            c3 = src[tid + 768];
        }

        // ---- A fragment: 16 px x 32 ch from padded NHWC activations
        const __bf16* arow =
            ynp + (((long long)b * PW_ + (h + kh)) * PW_ + (w0 + kw + m)) * C2_
                + c32 * 32 + koff;
        union { v16bf v; v8bf h2[2]; } A;
        A.h2[0] = *(const v8bf*)(arow);
        A.h2[1] = *(const v8bf*)(arow + 16);

        // ---- 16 WMMAs against LDS stage; swizzled, reg double-buffered reads
        const unsigned* bs = Bsm[s & 1];
        union { v16bf v; uint4 q[2]; } Bc, Bn;
        {
            const uint4* bq = (const uint4*)(bs + lane * 8);
            Bc.q[0] = bq[swz];                 // physical granule swz = logical 0
            Bc.q[1] = bq[swz ^ 1];             // physical granule !swz = logical 1
        }
#pragma unroll
        for (int ot = 0; ot < 16; ot++) {
            int otn = (ot + 1) & 15;
            const uint4* bq = (const uint4*)(bs + otn * 256 + lane * 8);
            Bn.q[0] = bq[swz];
            Bn.q[1] = bq[swz ^ 1];
            acc[ot] = wmma_bf16(A.v, Bc.v, acc[ot]);
            Bc.q[0] = Bn.q[0];
            Bc.q[1] = Bn.q[1];
        }

        // ---- commit next stage to the other LDS buffer
        if (s + 1 < 72) {
            uint4* dst = (uint4*)Bsm[(s + 1) & 1];
            dst[tid]       = c0;
            dst[tid + 256] = c1;
            dst[tid + 512] = c2;
            dst[tid + 768] = c3;
        }
        __syncthreads();
    }

    float a0 = attn[b * 4 + 0], a1 = attn[b * 4 + 1];
    float a2 = attn[b * 4 + 2], a3 = attn[b * 4 + 3];
    int n    = lane & 15;
    int prow = (lane & 16) ? 8 : 0;
#pragma unroll
    for (int ot = 0; ot < 16; ot++) {
        int o = ot * 16 + n;
        float ab = a0 * dyb[0 * C2_ + o] + a1 * dyb[1 * C2_ + o]
                 + a2 * dyb[2 * C2_ + o] + a3 * dyb[3 * C2_ + o];
        float s1 = b1g[o] * rsqrtf(b1v[o] + EPSV);
        float bi = (ab - b1m[o]) * s1 + b1b[o];
        float* op = out + ((long long)b * C2_ + o) * HW_ + p0 + prow;
        v4f lo, hi;
#pragma unroll
        for (int r = 0; r < 4; r++) { lo[r] = acc[ot][r] * s1 + bi;
                                      hi[r] = acc[ot][r + 4] * s1 + bi; }
        __builtin_nontemporal_store(lo, (v4f*)op);        // TH_NT: keep L2 for yn/W
        __builtin_nontemporal_store(hi, (v4f*)(op + 4));
    }
}

// ---------------------------------------------------------------------------
extern "C" void kernel_launch(void* const* d_in, const int* in_sizes, int n_in,
                              void* d_out, int out_size, void* d_ws, size_t ws_size,
                              hipStream_t stream) {
    const float* x    = (const float*)d_in[0];
    const float* w1   = (const float*)d_in[1];
    const float* bng  = (const float*)d_in[2];
    const float* bnb  = (const float*)d_in[3];
    const float* bnm  = (const float*)d_in[4];
    const float* bnv  = (const float*)d_in[5];
    const float* fc1w = (const float*)d_in[6];
    const float* fc1b = (const float*)d_in[7];
    const float* fc2w = (const float*)d_in[8];
    const float* fc2b = (const float*)d_in[9];
    const float* dyw  = (const float*)d_in[10];
    const float* dyb  = (const float*)d_in[11];
    const float* b1g  = (const float*)d_in[12];
    const float* b1b  = (const float*)d_in[13];
    const float* b1m  = (const float*)d_in[14];
    const float* b1v  = (const float*)d_in[15];
    (void)in_sizes; (void)n_in; (void)out_size; (void)ws_size;

    char* ws = (char*)d_ws;
    size_t off = 0;
    auto alloc = [&](size_t bytes) -> void* {
        void* p = ws + off;
        off += (bytes + 255) & ~(size_t)255;
        return p;
    };
    __bf16*   xn    = (__bf16*)  alloc((size_t)BN_ * HW_ * C1_ * 2);      // 26.2 MB
    unsigned* w1pk  = (unsigned*)alloc((size_t)16 * 4 * 32 * 8 * 4);      // 64 KB
    __bf16*   y     = (__bf16*)  alloc((size_t)BN_ * HW_ * C2_ * 2);      // 52.4 MB
    float*    gsum  = (float*)   alloc((size_t)BN_ * C2_ * 4);            // 16 KB
    float*    gsq   = (float*)   alloc((size_t)BN_ * C2_ * 4);
    float*    mu    = (float*)   alloc((size_t)BN_ * C2_ * 4);
    float*    rstd  = (float*)   alloc((size_t)BN_ * C2_ * 4);
    float*    attn  = (float*)   alloc(256);
    unsigned* aggpk = (unsigned*)alloc((size_t)BN_ * 9 * 8 * 16 * 32 * 8 * 4); // 18.9 MB
    __bf16*   ynp   = (__bf16*)  alloc((size_t)BN_ * PW_ * PW_ * C2_ * 2);     // 55.1 MB

    k_zero<<<dim3(16), dim3(256), 0, stream>>>((uint4*)gsum,
                                               (long long)(2 * BN_ * C2_ * 4) / 16);
    k_zero<<<dim3(2048), dim3(256), 0, stream>>>((uint4*)ynp,
                                                 ((long long)BN_ * PW_ * PW_ * C2_ * 2) / 16);
    long long xtotal = (long long)BN_ * C1_ * HW_;
    k_convert_x<<<dim3((unsigned)((xtotal + 255) / 256)), dim3(256), 0, stream>>>(x, xn);
    k_pack_w1<<<dim3(64), dim3(256), 0, stream>>>(w1, w1pk);
    k_conv1x1<<<dim3(50, 16), dim3(256), 0, stream>>>(xn, w1pk, bng, bnb, bnm, bnv,
                                                      y, gsum, gsq);
    k_finalize<<<dim3(16), dim3(256), 0, stream>>>(gsum, gsq, mu, rstd);
    k_attn<<<dim3(16), dim3(64), 0, stream>>>(mu, fc1w, fc1b, fc2w, fc2b, attn);
    k_agg<<<dim3(18432), dim3(256), 0, stream>>>(dyw, attn, aggpk);
    k_normpad<<<dim3(HW_, BN_), dim3(256), 0, stream>>>(y, mu, rstd, ynp);
    k_dyconv<<<dim3(50, 16), dim3(256), 0, stream>>>(ynp, aggpk, attn, dyb,
                                                     b1g, b1b, b1m, b1v,
                                                     (float*)d_out);
}
